// DMMFA_56307021250864
// MI455X (gfx1250) — matrix operand order repaired and checked by reading, not compiled, not run on hardware
//
#include <hip/hip_runtime.h>
#include <hip/hip_bf16.h>

typedef __attribute__((ext_vector_type(16))) __bf16 v16bf;
typedef __attribute__((ext_vector_type(8)))  float  v8f;

#define HEADS 8
#define DH 32
#define HW 128
#define NPIX 16384          // 128*128
#define CCH 256
#define BATCH 4

// ---------------------------------------------------------------------------
// fp32 -> bf16 hi/lo split (round-to-nearest-even), bit-exact and codegen-safe
// ---------------------------------------------------------------------------
__device__ __forceinline__ __bf16 f2bf(float x) {
    union { float f; unsigned u; } v; v.f = x;
    unsigned r = v.u + 0x7FFFu + ((v.u >> 16) & 1u);
    unsigned short hs = (unsigned short)(r >> 16);
    __bf16 out; __builtin_memcpy(&out, &hs, 2);
    return out;
}
__device__ __forceinline__ float bf2f(__bf16 b) {
    unsigned short hs; __builtin_memcpy(&hs, &b, 2);
    union { unsigned u; float f; } v; v.u = ((unsigned)hs) << 16;
    return v.f;
}
__device__ __forceinline__ void splitf(float x, __bf16& h, __bf16& l) {
    h = f2bf(x);
    l = f2bf(x - bf2f(h));
}

#define WMMA_BF16(a, b, c) \
    __builtin_amdgcn_wmma_f32_16x16x32_bf16(false, (a), false, (b), (short)0, (c), false, false)

// ---------------------------------------------------------------------------
// Kernel 1: bf16x3 split-precision GEMM  C[M,N] = A[M,K] * B[K,N]  (~fp32 acc)
// Block tile 64x64, 8 waves (4M x 2N), wave = 16x32 (two f32 accumulators).
// K staged in LDS in slabs of 32; hi/lo bf16 planes; B stored transposed.
// 3 bf16 WMMAs per 16x16x32 step: Ah*Bh + Ah*Bl + Al*Bh.
// ---------------------------------------------------------------------------
__global__ __launch_bounds__(256) void gemm_bf16x3_wmma(
    const float* __restrict__ A, const float* __restrict__ B,
    float* __restrict__ C, int M, int N, int K)
{
    __shared__ __bf16 lAh[64][38], lAl[64][38];   // [m][k], pad 38 halfs (19 words)
    __shared__ __bf16 lBh[64][38], lBl[64][38];   // TRANSPOSED [n][k]

    const int tid  = threadIdx.x;
    const int lane = tid & 31;
    const int wave = tid >> 5;
    const int half = lane >> 4;
    const int r    = lane & 15;
    const int waveM = wave >> 1;    // 0..3
    const int waveN = wave & 1;     // 0..1

    const int m0 = blockIdx.y * 64;
    const int n0 = blockIdx.x * 64;

    v8f c0 = {}; v8f c1 = {};

    const int aRow = tid >> 2;          // 0..63
    const int aCol = (tid & 3) * 8;     // 0,8,16,24
    const int bRow = tid >> 3;          // 0..31
    const int bCol = (tid & 7) * 8;     // 0..56

    for (int k0 = 0; k0 < K; k0 += 32) {
        // stage A slab 64x32 (split once per element)
        {
            const float* ap = A + (size_t)(m0 + aRow) * K + k0 + aCol;
            float4 a0 = *(const float4*)ap;
            float4 a1 = *(const float4*)(ap + 4);
            float av[8] = {a0.x, a0.y, a0.z, a0.w, a1.x, a1.y, a1.z, a1.w};
#pragma unroll
            for (int i = 0; i < 8; ++i) {
                __bf16 h, l; splitf(av[i], h, l);
                lAh[aRow][aCol + i] = h;
                lAl[aRow][aCol + i] = l;
            }
        }
        // stage B slab 32x64, transposed into [n][k]
        {
            const float* bp = B + (size_t)(k0 + bRow) * N + n0 + bCol;
            float4 b0 = *(const float4*)bp;
            float4 b1 = *(const float4*)(bp + 4);
            float bv[8] = {b0.x, b0.y, b0.z, b0.w, b1.x, b1.y, b1.z, b1.w};
#pragma unroll
            for (int i = 0; i < 8; ++i) {
                __bf16 h, l; splitf(bv[i], h, l);
                lBh[bCol + i][bRow] = h;
                lBl[bCol + i][bRow] = l;
            }
        }
        __syncthreads();

        // fragments (ISA 16-bit layouts)
        const int arow = waveM * 16 + r;
        v16bf ah, al;
#pragma unroll
        for (int j = 0; j < 16; ++j) {
            const int k = half * 8 + j + ((j >= 8) ? 8 : 0);
            ah[j] = lAh[arow][k];
            al[j] = lAl[arow][k];
        }
        const int bc0 = waveN * 32 + r;
        const int bc1 = bc0 + 16;
        v16bf bh0, bl0, bh1, bl1;
#pragma unroll
        for (int j = 0; j < 16; ++j) {
            const int k = half * 8 + j + ((j >= 8) ? 8 : 0);
            bh0[j] = lBh[bc0][k]; bl0[j] = lBl[bc0][k];
            bh1[j] = lBh[bc1][k]; bl1[j] = lBl[bc1][k];
        }

        c0 = WMMA_BF16(ah, bh0, c0);
        c0 = WMMA_BF16(ah, bl0, c0);
        c0 = WMMA_BF16(al, bh0, c0);
        c1 = WMMA_BF16(ah, bh1, c1);
        c1 = WMMA_BF16(ah, bl1, c1);
        c1 = WMMA_BF16(al, bh1, c1);
        __syncthreads();
    }

#pragma unroll
    for (int i = 0; i < 8; ++i) {
        const int row = m0 + waveM * 16 + i + 8 * half;
        const int col = n0 + waveN * 32 + r;
        C[(size_t)row * N + col]      = c0[i];
        C[(size_t)row * N + col + 16] = c1[i];
    }
}

// ---------------------------------------------------------------------------
// Kernel 2: 3x3 conv 256->16 as 9 accumulating bf16x3 WMMA GEMMs.
// Block = one image row (128 pixels); 8 waves, each a 16-pixel x 16-chan tile.
// NHWC rows are contiguous, so "im2col" = x-shifted row stage + border zeros.
// ---------------------------------------------------------------------------
__global__ __launch_bounds__(256) void off_conv_wmma(
    const float* __restrict__ x,     // [4,128,128,256] NHWC
    const float* __restrict__ wOff,  // [16,256,3,3] OIHW
    const float* __restrict__ bOff,  // [16]
    float* __restrict__ off)         // [4,128,128,16]
{
    __shared__ __bf16 lAh[128][38], lAl[128][38];  // [pixel][k]
    __shared__ __bf16 lBh[16][38],  lBl[16][38];   // TRANSPOSED [o][k]

    const int b = blockIdx.y;
    const int y = blockIdx.x;
    const int tid  = threadIdx.x;
    const int lane = tid & 31;
    const int wave = tid >> 5;      // 0..7 -> pixel tile
    const int half = lane >> 4;
    const int r    = lane & 15;

    v8f c = {};

    const int aRow = tid >> 1;         // 0..127 pixel
    const int aCol = (tid & 1) * 16;   // 0 or 16
    const int wo   = tid >> 4;         // 0..15 out channel
    const int wk   = (tid & 15) * 2;   // 0..30

    for (int ky = 0; ky < 3; ++ky) {
        const int yy = y + ky - 1;
        if (yy < 0 || yy >= HW) continue;   // block-uniform
        for (int kx = 0; kx < 3; ++kx) {
            const int xx  = aRow + kx - 1;
            const bool inb = (xx >= 0 && xx < HW);
            const float* xp = x + (((size_t)b * HW + yy) * HW + xx) * CCH + aCol;
            for (int k0 = 0; k0 < CCH; k0 += 32) {
                // stage weights W[o, k0+k, ky, kx] -> [o][k] hi/lo
#pragma unroll
                for (int i = 0; i < 2; ++i) {
                    const float wv =
                        wOff[(((size_t)wo * 256 + (k0 + wk + i)) * 3 + ky) * 3 + kx];
                    __bf16 h, l; splitf(wv, h, l);
                    lBh[wo][wk + i] = h;
                    lBl[wo][wk + i] = l;
                }
                // stage shifted image row slab 128x32 (zero at x borders)
#pragma unroll
                for (int i = 0; i < 16; i += 4) {
                    float4 v;
                    if (inb) v = *(const float4*)(xp + k0 + i);
                    else { v.x = 0.f; v.y = 0.f; v.z = 0.f; v.w = 0.f; }
                    __bf16 h, l;
                    splitf(v.x, h, l); lAh[aRow][aCol+i+0]=h; lAl[aRow][aCol+i+0]=l;
                    splitf(v.y, h, l); lAh[aRow][aCol+i+1]=h; lAl[aRow][aCol+i+1]=l;
                    splitf(v.z, h, l); lAh[aRow][aCol+i+2]=h; lAl[aRow][aCol+i+2]=l;
                    splitf(v.w, h, l); lAh[aRow][aCol+i+3]=h; lAl[aRow][aCol+i+3]=l;
                }
                __syncthreads();

                const int arow = wave * 16 + r;
                v16bf ah, al;
#pragma unroll
                for (int j = 0; j < 16; ++j) {
                    const int k = half * 8 + j + ((j >= 8) ? 8 : 0);
                    ah[j] = lAh[arow][k];
                    al[j] = lAl[arow][k];
                }
                v16bf bh, bl;
#pragma unroll
                for (int j = 0; j < 16; ++j) {
                    const int k = half * 8 + j + ((j >= 8) ? 8 : 0);
                    bh[j] = lBh[r][k];
                    bl[j] = lBl[r][k];
                }
                c = WMMA_BF16(ah, bh, c);
                c = WMMA_BF16(ah, bl, c);
                c = WMMA_BF16(al, bh, c);
                __syncthreads();
            }
        }
    }

    const float bias = bOff[r];
#pragma unroll
    for (int i = 0; i < 8; ++i) {
        const int p = wave * 16 + i + 8 * half;     // pixel (M), r = channel (N)
        off[(((size_t)b * HW + y) * HW + p) * 16 + r] = c[i] + bias;
    }
}

// ---------------------------------------------------------------------------
// Kernel 3: bilinear grid-sample (border, align_corners) + cosine logits.
// One wave per (b,head,pixel); lane = channel d -> coalesced 128B gathers.
// ---------------------------------------------------------------------------
__global__ __launch_bounds__(256) void sample_logit(
    const float* __restrict__ q,    // [4,16384,256]
    const float* __restrict__ kv,   // [4,16384,512] (k | v)
    const float* __restrict__ off,  // [4,128,128,16]
    float* __restrict__ logit,      // [32,16384]
    float* __restrict__ sv)         // [32,16384,32]
{
    const int lane = threadIdx.x & 31;
    const int item = blockIdx.x * 8 + (threadIdx.x >> 5);
    const int bh  = item >> 14;
    const int pix = item & (NPIX - 1);
    const int b = bh >> 3, head = bh & 7;
    const int y = pix >> 7, x = pix & 127;

    const float* op = off + (((size_t)b * HW + y) * HW + x) * 16 + head * 2;
    const float ox = op[0], oy = op[1];

    const float step = 2.0f / 127.0f;
    const float gx = (-1.0f + x * step) + ox;
    const float gy = (-1.0f + y * step) + oy;
    const float ix = fminf(fmaxf((gx + 1.0f) * 0.5f * 127.0f, 0.0f), 127.0f);
    const float iy = fminf(fmaxf((gy + 1.0f) * 0.5f * 127.0f, 0.0f), 127.0f);
    const float x0f = floorf(ix), y0f = floorf(iy);
    const int x0 = (int)x0f, y0 = (int)y0f;
    const int x1 = min(x0 + 1, 127), y1 = min(y0 + 1, 127);
    const float wx = ix - x0f, wy = iy - y0f;
    const float w00 = (1.0f - wx) * (1.0f - wy), w01 = wx * (1.0f - wy);
    const float w10 = (1.0f - wx) * wy,          w11 = wx * wy;

    const size_t base = (size_t)b * NPIX * 512 + head * DH + lane;
    const size_t i00 = base + (size_t)(y0 * HW + x0) * 512;
    const size_t i01 = base + (size_t)(y0 * HW + x1) * 512;
    const size_t i10 = base + (size_t)(y1 * HW + x0) * 512;
    const size_t i11 = base + (size_t)(y1 * HW + x1) * 512;

    const float skd = kv[i00] * w00 + kv[i01] * w01 + kv[i10] * w10 + kv[i11] * w11;
    const float svd = kv[i00 + 256] * w00 + kv[i01 + 256] * w01 +
                      kv[i10 + 256] * w10 + kv[i11 + 256] * w11;
    const float qd = q[((size_t)b * NPIX + pix) * CCH + head * DH + lane];

    float dq = qd * qd, ds = skd * skd, dd = qd * skd;
#pragma unroll
    for (int m = 16; m >= 1; m >>= 1) {
        dq += __shfl_xor(dq, m, 32);
        ds += __shfl_xor(ds, m, 32);
        dd += __shfl_xor(dd, m, 32);
    }
    if (lane == 0) {
        const float den = fmaxf(sqrtf(dq), 1e-12f) * fmaxf(sqrtf(ds), 1e-12f);
        logit[(size_t)bh * NPIX + pix] = dd / den;
    }
    sv[((size_t)bh * NPIX + pix) * DH + lane] = svd;
}

// ---------------------------------------------------------------------------
// Kernel 4: per-(b,head) softmax over 16384 logits + pooled = attn . sv.
// ---------------------------------------------------------------------------
__global__ __launch_bounds__(256) void softmax_pool(
    float* __restrict__ logit,      // [32,16384] (overwritten with exp)
    const float* __restrict__ sv,   // [32,16384,32]
    float* __restrict__ pooled)     // [4,256]
{
    __shared__ float red[256];
    __shared__ float pr[8][32];
    const int bh = blockIdx.x, t = threadIdx.x;
    float* L = logit + (size_t)bh * NPIX;

    float m = -3.0e38f;
    for (int n = t; n < NPIX; n += 256) m = fmaxf(m, L[n]);
    red[t] = m; __syncthreads();
    for (int s = 128; s > 0; s >>= 1) {
        if (t < s) red[t] = fmaxf(red[t], red[t + s]);
        __syncthreads();
    }
    const float mx = red[0]; __syncthreads();

    float sum = 0.0f;
    for (int n = t; n < NPIX; n += 256) {
        const float e = expf(L[n] - mx);
        L[n] = e; sum += e;
    }
    red[t] = sum; __syncthreads();
    for (int s = 128; s > 0; s >>= 1) {
        if (t < s) red[t] += red[t + s];
        __syncthreads();
    }
    const float denom = red[0]; __syncthreads();

    const int d = t & 31, g = t >> 5;
    const float* S = sv + (size_t)bh * NPIX * DH;
    float acc = 0.0f;
    for (int n = g; n < NPIX; n += 8)
        acc += L[n] * S[(size_t)n * DH + d];
    pr[g][d] = acc; __syncthreads();
    if (g == 0) {
        float s = 0.0f;
#pragma unroll
        for (int i = 0; i < 8; ++i) s += pr[i][d];
        const int b = bh >> 3, head = bh & 7;
        pooled[b * 256 + head * DH + d] = s / denom;
    }
}

// ---------------------------------------------------------------------------
// Kernel 5: row[c] = (pooled[b].w_proj[:,c] + b_proj[c])*scale[c], broadcast.
// ---------------------------------------------------------------------------
__global__ __launch_bounds__(256) void proj_broadcast(
    const float* __restrict__ pooled,  // [4,256]
    const float* __restrict__ wProj,   // [256,256]
    const float* __restrict__ bProj,   // [256]
    const float* __restrict__ lScale,  // [256]
    float* __restrict__ out)           // [4,16384,256]
{
    const int b = blockIdx.y;
    const int chunk = blockIdx.x;      // 0..63, 256 rows each
    const int c = threadIdx.x;

    const float* p = pooled + b * 256;
    float acc = bProj[c];
    for (int i = 0; i < 256; ++i)
        acc += p[i] * wProj[(size_t)i * 256 + c];
    acc *= lScale[c];

    float* op = out + ((size_t)b * NPIX + chunk * 256) * CCH + c;
    for (int n = 0; n < 256; ++n)
        op[(size_t)n * CCH] = acc;
}

// ---------------------------------------------------------------------------
extern "C" void kernel_launch(void* const* d_in, const int* in_sizes, int n_in,
                              void* d_out, int out_size, void* d_ws, size_t ws_size,
                              hipStream_t stream) {
    const float* x_in    = (const float*)d_in[0];
    const float* w_q     = (const float*)d_in[1];
    const float* w_kv    = (const float*)d_in[2];
    const float* w_proj  = (const float*)d_in[3];
    const float* b_proj  = (const float*)d_in[4];
    const float* w_off   = (const float*)d_in[5];
    const float* b_off   = (const float*)d_in[6];
    const float* l_scale = (const float*)d_in[7];
    float* out = (float*)d_out;

    const size_t M = (size_t)BATCH * NPIX;   // 65536
    float* q      = (float*)d_ws;            // M*256
    float* kv     = q + M * 256;             // M*512
    float* off    = kv + M * 512;            // 4*16384*16
    float* logit  = off + (size_t)BATCH * NPIX * 16;   // 32*16384
    float* sv     = logit + (size_t)32 * NPIX;         // 32*16384*32
    float* pooled = sv + (size_t)32 * NPIX * DH;       // 4*256

    // q = x @ w_q        (65536 x 256 x 256), bf16x3 split precision
    gemm_bf16x3_wmma<<<dim3(256 / 64, M / 64), 256, 0, stream>>>(
        x_in, w_q, q, (int)M, 256, 256);
    // kv = x @ w_kv      (65536 x 256 x 512)
    gemm_bf16x3_wmma<<<dim3(512 / 64, M / 64), 256, 0, stream>>>(
        x_in, w_kv, kv, (int)M, 512, 256);
    // deformable offsets: 3x3 conv as 9 accumulating WMMA GEMMs
    off_conv_wmma<<<dim3(HW, BATCH), 256, 0, stream>>>(
        x_in, w_off, b_off, off);
    // bilinear sample + cosine logits
    sample_logit<<<(BATCH * HEADS * NPIX) / 8, 256, 0, stream>>>(
        q, kv, off, logit, sv);
    // softmax + attention pooling
    softmax_pool<<<BATCH * HEADS, 256, 0, stream>>>(logit, sv, pooled);
    // projection + layer-scale + broadcast over n
    proj_broadcast<<<dim3(64, BATCH), 256, 0, stream>>>(
        pooled, w_proj, b_proj, l_scale, out);
}
